// AWD_LSTM_60662118088723
// MI455X (gfx1250) — matrix operand order repaired and checked by reading, not compiled, not run on hardware
//
#include <hip/hip_runtime.h>
#include <hip/hip_bf16.h>

typedef __attribute__((ext_vector_type(16))) __bf16 v16bf;
typedef __attribute__((ext_vector_type(8)))  float  v8f;

constexpr int E = 400, H = 1150, V = 10000, T = 256, B = 32;
constexpr int EP = 416, HP = 1152;          // K dims padded to mult of 32
constexpr int JT_H = 72, JT_E = 25, JT_V = 625;   // 16-row tiles per gate/output
constexpr int KT_E = 13, KT_H = 36;               // 32-wide K tiles

__device__ __forceinline__ float sigmoidf_(float x) {
    return 1.0f / (1.0f + __expf(-x));
}

__device__ __forceinline__ v8f wmma_bf16(v16bf a, v16bf b, v8f c) {
    return __builtin_amdgcn_wmma_f32_16x16x32_bf16(false, a, false, b, (short)0, c,
                                                   false, false);
}

// ---------------------------------------------------------------------------
// Pack fp32 weight [G*out_sz, K] into bf16 WMMA A-fragments.
// Tile (g, jt, kt) is 16 rows x 32 K, stored as 32 lanes x 16 halves so the
// GEMM kernel loads each lane's fragment with two contiguous b128 loads.
// 16-bit A layout used here (contiguous k, B matches; not ISA table order): lane = m (+16 for K-high-half),
//   half h -> K = h + (lane>=16 ? 16 : 0)
// Out-of-range rows/cols are zero-padded.
// ---------------------------------------------------------------------------
__global__ void pack_w(const float* __restrict__ W, __bf16* __restrict__ out,
                       int out_sz, int K, int G, int JT, int KT) {
    size_t total = (size_t)G * JT * KT * 512;
    for (size_t idx = (size_t)blockIdx.x * blockDim.x + threadIdx.x; idx < total;
         idx += (size_t)gridDim.x * blockDim.x) {
        int h    = (int)(idx & 15);
        int lane = (int)((idx >> 4) & 31);
        size_t tile = idx >> 9;
        int kt = (int)(tile % KT);
        size_t gj = tile / KT;
        int jt = (int)(gj % JT);
        int g  = (int)(gj / JT);
        int m = lane & 15;
        int k = h + ((lane >= 16) ? 16 : 0);
        int row = jt * 16 + m;
        int col = kt * 32 + k;
        float v = 0.0f;
        if (row < out_sz && col < K)
            v = W[(size_t)(g * out_sz + row) * K + col];
        out[idx] = (__bf16)v;
    }
}

// Embedding gather -> bf16 padded rows [T*B, EP]
__global__ void embed_gather(const int* __restrict__ x, const float* __restrict__ embW,
                             __bf16* __restrict__ out) {
    size_t total = (size_t)T * B * EP;
    for (size_t idx = (size_t)blockIdx.x * blockDim.x + threadIdx.x; idx < total;
         idx += (size_t)gridDim.x * blockDim.x) {
        int e = (int)(idx % EP);
        size_t r = idx / EP;
        float v = 0.0f;
        if (e < E) v = embW[(size_t)x[r] * E + e];
        out[idx] = (__bf16)v;
    }
}

__global__ void zero_bf16(__bf16* __restrict__ dst, int n) {
    for (int i = blockIdx.x * blockDim.x + threadIdx.x; i < n;
         i += gridDim.x * blockDim.x)
        dst[i] = (__bf16)0.0f;
}

__global__ void copy_f32(float* __restrict__ dst, const float* __restrict__ src, int n) {
    for (int i = blockIdx.x * blockDim.x + threadIdx.x; i < n;
         i += gridDim.x * blockDim.x)
        dst[i] = src[i];
}

__global__ void f32_to_bf16_pad(__bf16* __restrict__ dst, const float* __restrict__ src,
                                int n, int npad, int rows) {
    int total = rows * npad;
    for (int i = blockIdx.x * blockDim.x + threadIdx.x; i < total;
         i += gridDim.x * blockDim.x) {
        int e = i % npad;
        int r = i / npad;
        dst[i] = (e < n) ? (__bf16)src[(size_t)r * n + e] : (__bf16)0.0f;
    }
}

// ---------------------------------------------------------------------------
// One LSTM cell step. Grid = JT workgroups of 256 threads (8 wave32).
// Wave w handles gate g = w>>1 (i,f,o,g order) and batch half w&1.
// acc = Wi_tile x x^T + Wh_tile x h^T  via v_wmma_f32_16x16x32_bf16,
// then gates via LDS exchange. c (fp32) updated in place; h written as fp32
// (for final state output) and bf16 (padded, for next GEMMs).
// ---------------------------------------------------------------------------
template <int KTX, int KTH, int OUT, int JT, int LDX, int LDH, int LDHO>
__global__ __launch_bounds__(256) void lstm_cell(
    const __bf16* __restrict__ pWi, const __bf16* __restrict__ pWh,
    const __bf16* __restrict__ xb, const __bf16* __restrict__ hb,
    const float* __restrict__ bi, const float* __restrict__ bhs,
    float* __restrict__ cf, float* __restrict__ hf, __bf16* __restrict__ hb_out) {
    __shared__ float pacc[4][16][33];
    const int jt = blockIdx.x;
    const int w = threadIdx.x >> 5;
    const int lane = threadIdx.x & 31;
    const int g = w >> 1;
    const int nh = w & 1;
    const int hi16 = (lane >> 4) << 4;          // 0 or 16: K half select for B frag
    const int brow = nh * 16 + (lane & 15);     // batch row this lane supplies

    v8f acc = {};
    {
        const __bf16* pA = pWi + ((size_t)(g * JT + jt) * KTX) * 512 + lane * 16;
        const __bf16* pB = xb + (size_t)brow * LDX + hi16;
#pragma unroll 4
        for (int kt = 0; kt < KTX; ++kt) {
            v16bf a = *(const v16bf*)(pA + (size_t)kt * 512);
            v16bf b = *(const v16bf*)(pB + kt * 32);
            acc = wmma_bf16(a, b, acc);
        }
    }
    {
        const __bf16* pA = pWh + ((size_t)(g * JT + jt) * KTH) * 512 + lane * 16;
        const __bf16* pB = hb + (size_t)brow * LDH + hi16;
#pragma unroll 4
        for (int kt = 0; kt < KTH; ++kt) {
            v16bf a = *(const v16bf*)(pA + (size_t)kt * 512);
            v16bf b = *(const v16bf*)(pB + kt * 32);
            acc = wmma_bf16(a, b, acc);
        }
    }
    // C/D layout: VGPR v -> M = v (+8 for lanes>=16), N = lane&15
    const int mbase = (lane < 16) ? 0 : 8;
    const int bcol = nh * 16 + (lane & 15);
#pragma unroll
    for (int v = 0; v < 8; ++v) pacc[g][mbase + v][bcol] = acc[v];
    __syncthreads();

    for (int p = threadIdx.x; p < 512; p += 256) {
        int m = p >> 5;
        int b = p & 31;
        int j = jt * 16 + m;
        if (j < OUT) {
            float pi = pacc[0][m][b] + bi[0 * OUT + j] + bhs[0 * OUT + j];
            float pf = pacc[1][m][b] + bi[1 * OUT + j] + bhs[1 * OUT + j];
            float po = pacc[2][m][b] + bi[2 * OUT + j] + bhs[2 * OUT + j];
            float pg = pacc[3][m][b] + bi[3 * OUT + j] + bhs[3 * OUT + j];
            float ig = sigmoidf_(pi);
            float fg = sigmoidf_(pf);
            float og = sigmoidf_(po);
            float gg = tanhf(pg);
            size_t sidx = (size_t)b * OUT + j;
            float cn = fg * cf[sidx] + ig * gg;
            float hn = og * tanhf(cn);
            cf[sidx] = cn;
            hf[sidx] = hn;
            hb_out[(size_t)b * LDHO + j] = (__bf16)hn;
        }
    }
}

// ---------------------------------------------------------------------------
// Decoder: [8192, 400] x [400, 10000]^T + bias. Wave tile: 16 vocab x 16 seq.
// grid.x = 256 (N blocks of 32), grid.y = 157 (M blocks of 64).
// ---------------------------------------------------------------------------
__global__ __launch_bounds__(256) void decoder(
    const __bf16* __restrict__ pW, const __bf16* __restrict__ seq,
    const float* __restrict__ bias, float* __restrict__ out) {
    const int w = threadIdx.x >> 5;
    const int lane = threadIdx.x & 31;
    const int mt = blockIdx.y * 4 + (w >> 1);
    const int nt = blockIdx.x * 2 + (w & 1);
    if (mt >= JT_V) return;  // wave-uniform: EXEC stays all-1s for WMMA waves
    v8f acc = {};
    const __bf16* pA = pW + ((size_t)mt * KT_E) * 512 + lane * 16;
    const int nrow = nt * 16 + (lane & 15);
    const __bf16* pB = seq + (size_t)nrow * EP + ((lane >> 4) << 4);
#pragma unroll
    for (int kt = 0; kt < KT_E; ++kt) {
        v16bf a = *(const v16bf*)(pA + (size_t)kt * 512);
        v16bf b = *(const v16bf*)(pB + kt * 32);
        acc = wmma_bf16(a, b, acc);
    }
    const int mbase = (lane < 16) ? 0 : 8;
    const int v0 = mt * 16 + mbase;
    const int n = nt * 16 + (lane & 15);
#pragma unroll
    for (int v = 0; v < 8; ++v) acc[v] += bias[v0 + v];
    *(v8f*)(out + (size_t)n * V + v0) = acc;  // 32B-aligned contiguous store
}

// Final states: decoded | h01 | h2 | c01 | c2
__global__ void write_states(float* __restrict__ out, const float* __restrict__ h01f,
                             const float* __restrict__ h2f,
                             const float* __restrict__ c01f,
                             const float* __restrict__ c2f) {
    constexpr size_t DEC = (size_t)T * B * V;
    constexpr int NH = 2 * B * H;
    constexpr int NE = B * E;
    int total = 2 * NH + 2 * NE;
    for (int idx = blockIdx.x * blockDim.x + threadIdx.x; idx < total;
         idx += gridDim.x * blockDim.x) {
        float v;
        if (idx < NH)               v = h01f[idx];
        else if (idx < NH + NE)     v = h2f[idx - NH];
        else if (idx < 2 * NH + NE) v = c01f[idx - NH - NE];
        else                        v = c2f[idx - 2 * NH - NE];
        out[DEC + idx] = v;
    }
}

extern "C" void kernel_launch(void* const* d_in, const int* in_sizes, int n_in,
                              void* d_out, int out_size, void* d_ws, size_t ws_size,
                              hipStream_t stream) {
    (void)in_sizes; (void)n_in; (void)out_size; (void)ws_size;
    const int*   x    = (const int*)d_in[0];
    const float* h0   = (const float*)d_in[1];
    const float* c0   = (const float*)d_in[2];
    const float* h0l  = (const float*)d_in[3];
    const float* c0l  = (const float*)d_in[4];
    const float* embW = (const float*)d_in[5];
    const float* Wi0  = (const float*)d_in[6];
    const float* bi0  = (const float*)d_in[7];
    const float* Wh0  = (const float*)d_in[8];
    const float* bh0  = (const float*)d_in[9];
    const float* Wi1  = (const float*)d_in[10];
    const float* bi1  = (const float*)d_in[11];
    const float* Wh1  = (const float*)d_in[12];
    const float* bh1  = (const float*)d_in[13];
    const float* Wi2  = (const float*)d_in[14];
    const float* bi2  = (const float*)d_in[15];
    const float* Wh2  = (const float*)d_in[16];
    const float* bh2  = (const float*)d_in[17];
    const float* decW = (const float*)d_in[18];
    const float* decb = (const float*)d_in[19];
    float* out = (float*)d_out;

    char* ws = (char*)d_ws;
    size_t off = 0;
    auto alloc = [&](size_t bytes) -> char* {
        char* p = ws + off;
        off = (off + bytes + 255) & ~(size_t)255;
        return p;
    };
    __bf16* pW0i = (__bf16*)alloc((size_t)4 * JT_H * KT_E * 512 * 2);
    __bf16* pW0h = (__bf16*)alloc((size_t)4 * JT_H * KT_H * 512 * 2);
    __bf16* pW1i = (__bf16*)alloc((size_t)4 * JT_H * KT_H * 512 * 2);
    __bf16* pW1h = (__bf16*)alloc((size_t)4 * JT_H * KT_H * 512 * 2);
    __bf16* pW2i = (__bf16*)alloc((size_t)4 * JT_E * KT_H * 512 * 2);
    __bf16* pW2h = (__bf16*)alloc((size_t)4 * JT_E * KT_E * 512 * 2);
    __bf16* pDec = (__bf16*)alloc((size_t)JT_V * KT_E * 512 * 2);
    __bf16* embB = (__bf16*)alloc((size_t)T * B * EP * 2);
    __bf16* seqB = (__bf16*)alloc((size_t)T * B * EP * 2);
    __bf16* h0bA = (__bf16*)alloc((size_t)B * HP * 2);
    __bf16* h0bB = (__bf16*)alloc((size_t)B * HP * 2);
    __bf16* h1bA = (__bf16*)alloc((size_t)B * HP * 2);
    __bf16* h1bB = (__bf16*)alloc((size_t)B * HP * 2);
    __bf16* h2bA = (__bf16*)alloc((size_t)B * EP * 2);
    float* h01f = (float*)alloc((size_t)2 * B * H * 4);
    float* c01f = (float*)alloc((size_t)2 * B * H * 4);
    float* h2f  = (float*)alloc((size_t)B * E * 4);
    float* c2f  = (float*)alloc((size_t)B * E * 4);

    // --- one-time-per-call weight packing + input prep (bf16, L2-resident) ---
    pack_w<<<2048, 256, 0, stream>>>(Wi0, pW0i, H, E, 4, JT_H, KT_E);
    pack_w<<<2048, 256, 0, stream>>>(Wh0, pW0h, H, H, 4, JT_H, KT_H);
    pack_w<<<2048, 256, 0, stream>>>(Wi1, pW1i, H, H, 4, JT_H, KT_H);
    pack_w<<<2048, 256, 0, stream>>>(Wh1, pW1h, H, H, 4, JT_H, KT_H);
    pack_w<<<2048, 256, 0, stream>>>(Wi2, pW2i, E, H, 4, JT_E, KT_H);
    pack_w<<<2048, 256, 0, stream>>>(Wh2, pW2h, E, E, 4, JT_E, KT_E);
    pack_w<<<2048, 256, 0, stream>>>(decW, pDec, V, E, 1, JT_V, KT_E);

    embed_gather<<<2048, 256, 0, stream>>>(x, embW, embB);
    zero_bf16<<<2048, 256, 0, stream>>>(seqB, T * B * EP);
    zero_bf16<<<160, 256, 0, stream>>>(h0bB, B * HP);
    zero_bf16<<<160, 256, 0, stream>>>(h1bB, B * HP);

    copy_f32<<<288, 256, 0, stream>>>(h01f, h0, 2 * B * H);
    copy_f32<<<288, 256, 0, stream>>>(c01f, c0, 2 * B * H);
    copy_f32<<<64, 256, 0, stream>>>(h2f, h0l, B * E);
    copy_f32<<<64, 256, 0, stream>>>(c2f, c0l, B * E);
    f32_to_bf16_pad<<<160, 256, 0, stream>>>(h0bA, h0, H, HP, B);
    f32_to_bf16_pad<<<160, 256, 0, stream>>>(h1bA, h0 + (size_t)B * H, H, HP, B);
    f32_to_bf16_pad<<<64, 256, 0, stream>>>(h2bA, h0l, E, EP, B);

    // --- recurrence: 3 cell kernels per timestep, h bf16 ping-ponged ---
    __bf16* h0cur = h0bA; __bf16* h0nxt = h0bB;
    __bf16* h1cur = h1bA; __bf16* h1nxt = h1bB;
    const __bf16* h2cur = h2bA;
    for (int t = 0; t < T; ++t) {
        lstm_cell<KT_E, KT_H, H, JT_H, EP, HP, HP><<<JT_H, 256, 0, stream>>>(
            pW0i, pW0h, embB + (size_t)t * B * EP, h0cur, bi0, bh0,
            c01f, h01f, h0nxt);
        lstm_cell<KT_H, KT_H, H, JT_H, HP, HP, HP><<<JT_H, 256, 0, stream>>>(
            pW1i, pW1h, h0nxt, h1cur, bi1, bh1,
            c01f + (size_t)B * H, h01f + (size_t)B * H, h1nxt);
        __bf16* h2out = seqB + (size_t)t * B * EP;  // feeds decoder AND next step
        lstm_cell<KT_H, KT_E, E, JT_E, HP, EP, EP><<<JT_E, 256, 0, stream>>>(
            pW2i, pW2h, h1nxt, h2cur, bi2, bh2, c2f, h2f, h2out);
        h2cur = h2out;
        { __bf16* tmp = h0cur; h0cur = h0nxt; h0nxt = tmp; }
        { __bf16* tmp = h1cur; h1cur = h1nxt; h1nxt = tmp; }
    }

    // --- decoder + final states ---
    decoder<<<dim3(256, 157), 256, 0, stream>>>(pDec, seqB, decb, out);
    write_states<<<675, 256, 0, stream>>>(out, h01f, h2f, c01f, c2f);
}
